// GCN3D_segR_21603685499511
// MI455X (gfx1250) — matrix operand order repaired and checked by reading, not compiled, not run on hardware
//
#include <hip/hip_runtime.h>
#include <stdint.h>

#define SUPPORT 7
#define NBR 10

typedef __attribute__((ext_vector_type(16))) _Float16 v16h;
typedef __attribute__((ext_vector_type(8)))  _Float16 v8h;
typedef __attribute__((ext_vector_type(8)))  float    v8f;

union HVec { v16h v; v8h h[2]; };

// ---------------------------------------------------------------------------
// Async global->LDS 16B copy (gfx1250, ASYNCcnt-tracked). lds_off is the LDS
// byte offset (low 32 bits of the generic shared-memory address);
// gptr is the per-lane 64-bit global address.
// ---------------------------------------------------------------------------
__device__ __forceinline__ void async_copy_b128(uint32_t lds_off, const void* gptr) {
  asm volatile("global_load_async_to_lds_b128 %0, %1, off"
               :: "v"(lds_off), "v"(gptr) : "memory");
}

// ---------------------------------------------------------------------------
// Normalize columns of a (3 x K) direction matrix.
// ---------------------------------------------------------------------------
__global__ void normalize_cols3(const float* __restrict__ d, float* __restrict__ dn, int K) {
  int k = blockIdx.x * 256 + threadIdx.x;
  if (k < K) {
    float x = d[k], y = d[K + k], z = d[2 * K + k];
    float n = sqrtf(x * x + y * y + z * z);
    float inv = 1.0f / fmaxf(n, 1e-12f);
    dn[k] = x * inv; dn[K + k] = y * inv; dn[2 * K + k] = z * inv;
  }
}

// ---------------------------------------------------------------------------
// Brute-force KNN (excluding self), LDS-tiled, register insertion sort.
// grid = (V/128, B), block = 128
// ---------------------------------------------------------------------------
template <int KB>
__global__ void knn_kernel(const float* __restrict__ verts, int* __restrict__ outIdx, int V) {
  int b = blockIdx.y;
  int q = blockIdx.x * 128 + threadIdx.x;
  const float* vb = verts + (size_t)b * V * 3;
  float qx = vb[q * 3 + 0], qy = vb[q * 3 + 1], qz = vb[q * 3 + 2];

  float bestd[KB];
  int   besti[KB];
#pragma unroll
  for (int i = 0; i < KB; ++i) { bestd[i] = 3.4e38f; besti[i] = 0; }

  __shared__ float tile[128 * 3];
  for (int t0 = 0; t0 < V; t0 += 128) {
    int j = t0 + threadIdx.x;
    tile[threadIdx.x * 3 + 0] = vb[j * 3 + 0];
    tile[threadIdx.x * 3 + 1] = vb[j * 3 + 1];
    tile[threadIdx.x * 3 + 2] = vb[j * 3 + 2];
    __syncthreads();
    for (int jj = 0; jj < 128; ++jj) {
      int j2 = t0 + jj;
      if (j2 == q) continue;
      float dx = tile[jj * 3 + 0] - qx;
      float dy = tile[jj * 3 + 1] - qy;
      float dz = tile[jj * 3 + 2] - qz;
      float cd = dx * dx + dy * dy + dz * dz;
      if (cd < bestd[KB - 1]) {
        int ci = j2;
#pragma unroll
        for (int p = 0; p < KB; ++p) {
          if (cd < bestd[p]) {
            float td = bestd[p]; int ti = besti[p];
            bestd[p] = cd; besti[p] = ci;
            cd = td; ci = ti;
          }
        }
      }
    }
    __syncthreads();
  }
#pragma unroll
  for (int n = 0; n < KB; ++n)
    outIdx[((size_t)b * V + q) * KB + n] = besti[n];
}

// ---------------------------------------------------------------------------
// Surface conv: fm0[b,v,c] = relu( sum_s max_n relu(<dirnorm(b,v,n), dn[:,s*128+c]>) )
// grid = (V, B), block = 128
// ---------------------------------------------------------------------------
__global__ void conv_surface_kernel(const float* __restrict__ verts,
                                    const int* __restrict__ idx,
                                    const float* __restrict__ dn,   // 3 x (SUPPORT*128)
                                    float* __restrict__ out,        // B x V x 128
                                    int V) {
  int b = blockIdx.y, v = blockIdx.x;
  __shared__ float nb[NBR][3];
  const float* vb = verts + (size_t)b * V * 3;
  if (threadIdx.x < NBR) {
    int j = idx[((size_t)b * V + v) * NBR + threadIdx.x];
    float dx = vb[j * 3 + 0] - vb[v * 3 + 0];
    float dy = vb[j * 3 + 1] - vb[v * 3 + 1];
    float dz = vb[j * 3 + 2] - vb[v * 3 + 2];
    float inv = 1.0f / fmaxf(sqrtf(dx * dx + dy * dy + dz * dz), 1e-12f);
    nb[threadIdx.x][0] = dx * inv;
    nb[threadIdx.x][1] = dy * inv;
    nb[threadIdx.x][2] = dz * inv;
  }
  __syncthreads();
  int c = threadIdx.x;
  const int K = SUPPORT * 128;
  float acc = 0.0f;
#pragma unroll
  for (int s = 0; s < SUPPORT; ++s) {
    int col = s * 128 + c;
    float wx = dn[col], wy = dn[K + col], wz = dn[2 * K + col];
    float m = 0.0f;  // relu'd thetas are >= 0, so 0 is the identity for max
#pragma unroll
    for (int n = 0; n < NBR; ++n)
      m = fmaxf(m, nb[n][0] * wx + nb[n][1] * wy + nb[n][2] * wz);
    acc += m;
  }
  out[((size_t)b * V + v) * 128 + c] = fmaxf(acc, 0.0f);
}

// ---------------------------------------------------------------------------
// f32 -> f16 convert (A matrix), and transposed convert (W: KxN -> Wt: NxK).
// ---------------------------------------------------------------------------
__global__ void cvt_f32_f16(const float* __restrict__ src, _Float16* __restrict__ dst, int n) {
  int i = blockIdx.x * 256 + threadIdx.x;
  if (i < n) dst[i] = (_Float16)src[i];
}

__global__ void transpose_f32_f16(const float* __restrict__ W, _Float16* __restrict__ Wt,
                                  int K, int N) {
  int i = blockIdx.x * 256 + threadIdx.x;
  if (i < K * N) {
    int k = i / N, n = i - k * N;
    Wt[(size_t)n * K + k] = (_Float16)W[i];
  }
}

// ---------------------------------------------------------------------------
// WMMA GEMM with async double-buffered LDS staging:
//   F[M,N] = A16[M,K] * Wt16[N,K]^T + bias[N], f32 accumulate.
// Block = 256 threads (8 waves) computing a 32(M) x 64(N) tile; wave w owns
// the 16x16 subtile (mw = w&1, nw = w>>1). K staged in 32-wide chunks through
// a double-buffered LDS tile filled by global_load_async_to_lds_b128:
//   A stage: 32 rows x 64B = 128 lanes (waves 0-3 issue A+B -> 2 async/stage)
//   B stage: 64 rows x 64B = 256 lanes (waves 4-7 issue B only -> 1 async/stage)
// s_wait_asynccnt with threshold 2/1 leaves the next stage's copies in flight.
// grid = (N/64, M/32), block = 256.
// ---------------------------------------------------------------------------
__global__ void wmma_gemm_bias(const _Float16* __restrict__ A16,   // M x K
                               const _Float16* __restrict__ Wt16,  // N x K
                               const float* __restrict__ bias,     // N
                               float* __restrict__ F,              // M x N
                               int M, int N, int K) {
  __shared__ __align__(16) _Float16 As[2][32][32];
  __shared__ __align__(16) _Float16 Bs[2][64][32];

  const int nt0 = blockIdx.x * 64;
  const int mt0 = blockIdx.y * 32;
  const int wave = threadIdx.x >> 5;
  const int lane = threadIdx.x & 31;
  const int lrow = lane & 15;
  const int khi  = lane >> 4;
  const int mw = wave & 1;
  const int nw = wave >> 1;

  auto issue_stage = [&](int buf, int kk) {
    {   // B tile: all 256 threads, one 16B chunk each
      int row = threadIdx.x >> 2;   // 0..63
      int seg = threadIdx.x & 3;    // 0..3
      const _Float16* g = Wt16 + (size_t)(nt0 + row) * K + kk + seg * 8;
      async_copy_b128((uint32_t)(uintptr_t)&Bs[buf][row][seg * 8], g);
    }
    if (threadIdx.x < 128) {  // A tile: waves 0-3
      int row = threadIdx.x >> 2;   // 0..31
      int seg = threadIdx.x & 3;
      const _Float16* g = A16 + (size_t)(mt0 + row) * K + kk + seg * 8;
      async_copy_b128((uint32_t)(uintptr_t)&As[buf][row][seg * 8], g);
    }
  };

  v8f c = {};
  const int nstage = K >> 5;
  issue_stage(0, 0);

  for (int s = 0; s < nstage; ++s) {
    int buf = s & 1;
    if (s + 1 < nstage) {
      issue_stage(buf ^ 1, (s + 1) << 5);
      // wait for current stage only; next stage stays in flight (in-order done)
      if (threadIdx.x < 128) asm volatile("s_wait_asynccnt 0x2" ::: "memory");
      else                   asm volatile("s_wait_asynccnt 0x1" ::: "memory");
    } else {
      asm volatile("s_wait_asynccnt 0x0" ::: "memory");
    }
    __syncthreads();

    HVec a, bfrag;
    const _Float16* ar = &As[buf][mw * 16 + lrow][0];
    const _Float16* br = &Bs[buf][nw * 16 + lrow][0];
    a.h[0]     = *(const v8h*)(ar + khi * 8);
    a.h[1]     = *(const v8h*)(ar + 16 + khi * 8);
    bfrag.h[0] = *(const v8h*)(br + khi * 8);
    bfrag.h[1] = *(const v8h*)(br + 16 + khi * 8);
    c = __builtin_amdgcn_wmma_f32_16x16x32_f16(
        /*neg_a=*/false, a.v, /*neg_b=*/false, bfrag.v,
        /*c_mod=*/(short)0, c, /*reuse_a=*/false, /*reuse_b=*/false);

    __syncthreads();  // buffer free for stage s+2 before its async writes land
  }

  const int mt = mt0 + mw * 16;
  const int nt = nt0 + nw * 16;
  float bb = bias[nt + lrow];
#pragma unroll
  for (int r = 0; r < 8; ++r) {
    int m = mt + r + 8 * khi;
    F[(size_t)m * N + nt + lrow] = c[r] + bb;
  }
}

// ---------------------------------------------------------------------------
// Conv-layer epilogue: out[b,v,c] = f_center + sum_s max_n theta(n,s,c)*support(nb_n,s,c)
// grid = (V, B), block = outC
// ---------------------------------------------------------------------------
__global__ void conv_layer_post(const float* __restrict__ verts,  // B x V x 3
                                const int* __restrict__ idx,      // B x V x NBR
                                const float* __restrict__ dn,     // 3 x (SUPPORT*outC)
                                const float* __restrict__ f,      // (B*V) x ((SUPPORT+1)*outC)
                                float* __restrict__ out,          // B x V x outC
                                int V, int outC) {
  int b = blockIdx.y, v = blockIdx.x;
  int c = threadIdx.x;
  __shared__ float nb[NBR][3];
  __shared__ int   nbi[NBR];
  const float* vb = verts + (size_t)b * V * 3;
  const int Ncol = (SUPPORT + 1) * outC;
  if (c < NBR) {
    int j = idx[((size_t)b * V + v) * NBR + c];
    nbi[c] = j;
    __builtin_prefetch(f + ((size_t)b * V + j) * Ncol + outC, 0, 0);
    float dx = vb[j * 3 + 0] - vb[v * 3 + 0];
    float dy = vb[j * 3 + 1] - vb[v * 3 + 1];
    float dz = vb[j * 3 + 2] - vb[v * 3 + 2];
    float inv = 1.0f / fmaxf(sqrtf(dx * dx + dy * dy + dz * dz), 1e-12f);
    nb[c][0] = dx * inv; nb[c][1] = dy * inv; nb[c][2] = dz * inv;
  }
  __syncthreads();

  const int K = SUPPORT * outC;
  float acc = f[((size_t)b * V + v) * Ncol + c];  // center
#pragma unroll
  for (int s = 0; s < SUPPORT; ++s) {
    int col = s * outC + c;
    float wx = dn[col], wy = dn[K + col], wz = dn[2 * K + col];
    float m = -3.4e38f;
#pragma unroll
    for (int n = 0; n < NBR; ++n) {
      float th = fmaxf(nb[n][0] * wx + nb[n][1] * wy + nb[n][2] * wz, 0.0f);
      float sv = f[((size_t)b * V + nbi[n]) * Ncol + outC + col];
      m = fmaxf(m, th * sv);
    }
    acc += m;
  }
  out[((size_t)b * V + v) * outC + c] = acc;
}

// ---------------------------------------------------------------------------
// BatchNorm stats (population mean/var over M rows per channel) + apply+relu.
// ---------------------------------------------------------------------------
__global__ void bn_stats(const float* __restrict__ x, int M, int C,
                         float* __restrict__ mean, float* __restrict__ rstd) {
  int c = blockIdx.x;
  __shared__ float ssum[256], ssq[256];
  float s = 0.0f, q = 0.0f;
  for (int r = threadIdx.x; r < M; r += 256) {
    float v = x[(size_t)r * C + c];
    s += v; q += v * v;
  }
  ssum[threadIdx.x] = s; ssq[threadIdx.x] = q;
  __syncthreads();
  for (int o = 128; o > 0; o >>= 1) {
    if (threadIdx.x < (unsigned)o) {
      ssum[threadIdx.x] += ssum[threadIdx.x + o];
      ssq[threadIdx.x]  += ssq[threadIdx.x + o];
    }
    __syncthreads();
  }
  if (threadIdx.x == 0) {
    float m = ssum[0] / (float)M;
    float var = fmaxf(ssq[0] / (float)M - m * m, 0.0f);
    mean[c] = m;
    rstd[c] = rsqrtf(var + 1e-5f);
  }
}

__global__ void bn_apply_relu(const float* __restrict__ x,
                              const float* __restrict__ mean, const float* __restrict__ rstd,
                              const float* __restrict__ g, const float* __restrict__ be,
                              float* __restrict__ y, int total, int C) {
  int i = blockIdx.x * 256 + threadIdx.x;
  if (i < total) {
    int c = i % C;
    float v = g[c] * (x[i] - mean[c]) * rstd[c] + be[c];
    y[i] = fmaxf(v, 0.0f);
  }
}

// ---------------------------------------------------------------------------
// Pool: deterministic stride permutation select + 4-NN max pool.
// grid = (pn, B), block = 128
// ---------------------------------------------------------------------------
__global__ void pool_kernel(const float* __restrict__ verts, const float* __restrict__ fm,
                            const int* __restrict__ idxp,  // B x V x 4
                            float* __restrict__ vout,      // B x pn x 3
                            float* __restrict__ fout,      // B x pn x C
                            int V, int pn, int C, int mult, int off) {
  int b = blockIdx.y, i = blockIdx.x;
  int sv = (i * mult + off) & (V - 1);  // V is a power of two; odd mult -> bijection
  if (threadIdx.x < 3)
    vout[((size_t)b * pn + i) * 3 + threadIdx.x] = verts[((size_t)b * V + sv) * 3 + threadIdx.x];
  int j0 = idxp[((size_t)b * V + sv) * 4 + 0];
  int j1 = idxp[((size_t)b * V + sv) * 4 + 1];
  int j2 = idxp[((size_t)b * V + sv) * 4 + 2];
  int j3 = idxp[((size_t)b * V + sv) * 4 + 3];
  for (int c = threadIdx.x; c < C; c += blockDim.x) {
    float m = fm[((size_t)b * V + j0) * C + c];
    m = fmaxf(m, fm[((size_t)b * V + j1) * C + c]);
    m = fmaxf(m, fm[((size_t)b * V + j2) * C + c]);
    m = fmaxf(m, fm[((size_t)b * V + j3) * C + c]);
    fout[((size_t)b * pn + i) * C + c] = m;
  }
}

// ---------------------------------------------------------------------------
// Orchestration
// ---------------------------------------------------------------------------
extern "C" void kernel_launch(void* const* d_in, const int* in_sizes, int n_in,
                              void* d_out, int out_size, void* d_ws, size_t ws_size,
                              hipStream_t stream) {
  (void)in_sizes; (void)n_in; (void)out_size; (void)ws_size;

  const float* verts = (const float*)d_in[0];
  const float* d0  = (const float*)d_in[1];
  const float* w1  = (const float*)d_in[2];
  const float* b1  = (const float*)d_in[3];
  const float* d1  = (const float*)d_in[4];
  const float* g1  = (const float*)d_in[5];
  const float* be1 = (const float*)d_in[6];
  const float* w2  = (const float*)d_in[7];
  const float* b2  = (const float*)d_in[8];
  const float* d2  = (const float*)d_in[9];
  const float* g2  = (const float*)d_in[10];
  const float* be2 = (const float*)d_in[11];
  const float* w3  = (const float*)d_in[12];
  const float* b3  = (const float*)d_in[13];
  const float* d3  = (const float*)d_in[14];
  const float* g3  = (const float*)d_in[15];
  const float* be3 = (const float*)d_in[16];
  const float* w4  = (const float*)d_in[17];
  const float* b4  = (const float*)d_in[18];
  const float* d4  = (const float*)d_in[19];

  const int B = 4, V = 2048, V1 = 512, V2 = 128;

  float* out  = (float*)d_out;
  float* fm0  = out;                 // 4*2048*128
  float* fm1  = out + 1048576;       // 4*2048*128
  float* vp1  = out + 2097152;       // 4*512*3
  float* fm2  = out + 2103296;       // 4*512*256
  float* fm3  = out + 2627584;       // 4*512*256
  float* vp2  = out + 3151872;       // 4*128*3
  float* fm4  = out + 3153408;       // 4*128*512

  uint8_t* base = (uint8_t*)d_ws;
  size_t cur = 0;
  auto take = [&](size_t bytes) -> void* {
    void* p = base + cur;
    cur += (bytes + 255) & ~(size_t)255;
    return p;
  };

  float* dn0 = (float*)take(3 * 896  * 4);
  float* dn1 = (float*)take(3 * 896  * 4);
  float* dn2 = (float*)take(3 * 1792 * 4);
  float* dn3 = (float*)take(3 * 1792 * 4);
  float* dn4 = (float*)take(3 * 3584 * 4);
  int* idx1  = (int*)take((size_t)B * V  * NBR * 4);
  int* idxp1 = (int*)take((size_t)B * V  * 4   * 4);
  int* idx2  = (int*)take((size_t)B * V1 * NBR * 4);
  int* idxp2 = (int*)take((size_t)B * V1 * 4   * 4);
  int* idx3  = (int*)take((size_t)B * V2 * NBR * 4);
  float* fbuf = (float*)take((size_t)8192 * 1024 * 4);   // reused GEMM output
  _Float16* A16  = (_Float16*)take((size_t)8192 * 128 * 2);
  _Float16* Wt16 = (_Float16*)take((size_t)4096 * 256 * 2);
  float* raw  = (float*)take((size_t)8192 * 128 * 4);    // pre-BN conv output
  float* fmp1 = (float*)take((size_t)B * V1 * 128 * 4);
  float* fmp2 = (float*)take((size_t)B * V2 * 256 * 4);
  float* mean = (float*)take(512 * 4);
  float* rstd = (float*)take(512 * 4);

  // --- normalize all direction banks ---
  normalize_cols3<<<4,  256, 0, stream>>>(d0, dn0, 896);
  normalize_cols3<<<4,  256, 0, stream>>>(d1, dn1, 896);
  normalize_cols3<<<7,  256, 0, stream>>>(d2, dn2, 1792);
  normalize_cols3<<<7,  256, 0, stream>>>(d3, dn3, 1792);
  normalize_cols3<<<14, 256, 0, stream>>>(d4, dn4, 3584);

  // --- stage 0: knn + surface conv -> fm0 ---
  knn_kernel<NBR><<<dim3(V / 128, B), 128, 0, stream>>>(verts, idx1, V);
  conv_surface_kernel<<<dim3(V, B), 128, 0, stream>>>(verts, idx1, dn0, fm0, V);

  // --- layer 1: f = fm0 @ w1 + b1 ; conv ; bn ; relu -> fm1 ---
  cvt_f32_f16<<<4096, 256, 0, stream>>>(fm0, A16, 8192 * 128);
  transpose_f32_f16<<<512, 256, 0, stream>>>(w1, Wt16, 128, 1024);
  wmma_gemm_bias<<<dim3(1024 / 64, 8192 / 32), 256, 0, stream>>>(A16, Wt16, b1, fbuf, 8192, 1024, 128);
  conv_layer_post<<<dim3(V, B), 128, 0, stream>>>(verts, idx1, dn1, fbuf, raw, V, 128);
  bn_stats<<<128, 256, 0, stream>>>(raw, 8192, 128, mean, rstd);
  bn_apply_relu<<<4096, 256, 0, stream>>>(raw, mean, rstd, g1, be1, fm1, 8192 * 128, 128);

  // --- pool 1: knn4 on verts, max pool fm1, select 512 -> vp1, fmp1 ---
  knn_kernel<4><<<dim3(V / 128, B), 128, 0, stream>>>(verts, idxp1, V);
  pool_kernel<<<dim3(V1, B), 128, 0, stream>>>(verts, fm1, idxp1, vp1, fmp1, V, V1, 128, 1117, 101);

  // --- layer 2 on vp1 ---
  knn_kernel<NBR><<<dim3(V1 / 128, B), 128, 0, stream>>>(vp1, idx2, V1);
  cvt_f32_f16<<<1024, 256, 0, stream>>>(fmp1, A16, 2048 * 128);
  transpose_f32_f16<<<1024, 256, 0, stream>>>(w2, Wt16, 128, 2048);
  wmma_gemm_bias<<<dim3(2048 / 64, 2048 / 32), 256, 0, stream>>>(A16, Wt16, b2, fbuf, 2048, 2048, 128);
  conv_layer_post<<<dim3(V1, B), 256, 0, stream>>>(vp1, idx2, dn2, fbuf, raw, V1, 256);
  bn_stats<<<256, 256, 0, stream>>>(raw, 2048, 256, mean, rstd);
  bn_apply_relu<<<2048, 256, 0, stream>>>(raw, mean, rstd, g2, be2, fm2, 2048 * 256, 256);

  // --- layer 3 on vp1 ---
  cvt_f32_f16<<<2048, 256, 0, stream>>>(fm2, A16, 2048 * 256);
  transpose_f32_f16<<<2048, 256, 0, stream>>>(w3, Wt16, 256, 2048);
  wmma_gemm_bias<<<dim3(2048 / 64, 2048 / 32), 256, 0, stream>>>(A16, Wt16, b3, fbuf, 2048, 2048, 256);
  conv_layer_post<<<dim3(V1, B), 256, 0, stream>>>(vp1, idx2, dn3, fbuf, raw, V1, 256);
  bn_stats<<<256, 256, 0, stream>>>(raw, 2048, 256, mean, rstd);
  bn_apply_relu<<<2048, 256, 0, stream>>>(raw, mean, rstd, g3, be3, fm3, 2048 * 256, 256);

  // --- pool 2: knn4 on vp1, max pool fm3, select 128 -> vp2, fmp2 ---
  knn_kernel<4><<<dim3(V1 / 128, B), 128, 0, stream>>>(vp1, idxp2, V1);
  pool_kernel<<<dim3(V2, B), 128, 0, stream>>>(vp1, fm3, idxp2, vp2, fmp2, V1, V2, 256, 733, 202);

  // --- layer 4 on vp2 (no bn/relu) -> fm4 ---
  knn_kernel<NBR><<<dim3(1, B), 128, 0, stream>>>(vp2, idx3, V2);
  cvt_f32_f16<<<512, 256, 0, stream>>>(fmp2, A16, 512 * 256);
  transpose_f32_f16<<<4096, 256, 0, stream>>>(w4, Wt16, 256, 4096);
  wmma_gemm_bias<<<dim3(4096 / 64, 512 / 32), 256, 0, stream>>>(A16, Wt16, b4, fbuf, 512, 4096, 256);
  conv_layer_post<<<dim3(V2, B), 512, 0, stream>>>(vp2, idx3, dn4, fbuf, fm4, V2, 512);
}
